// MultiHeadAttentionBlock_27419071217724
// MI455X (gfx1250) — matrix operand order repaired and checked by reading, not compiled, not run on hardware
//
#include <hip/hip_runtime.h>
#include <hip/hip_bf16.h>

// ---------------- CDNA5 WMMA types ----------------
typedef __attribute__((ext_vector_type(16))) __bf16 v16bf;
typedef __attribute__((ext_vector_type(8)))  __bf16 v8bf;
typedef __attribute__((ext_vector_type(8)))  float  v8f;

static constexpr int Bn  = 4;
static constexpr int Sn  = 2048;
static constexpr int Dn  = 1024;
static constexpr int Hn  = 16;
static constexpr int DKn = 64;
static constexpr int Rows = Bn * Sn;   // 8192

union AFragU { v16bf v; v8bf h[2]; };

// A-matrix 16x32 bf16 fragment (M x K), lane layout used here:
// lane m = lane&15, khalf = lane>>4; lane holds K = khalf*16 + {0..15}
// (contiguous per lane, matching the B fragment).
__device__ __forceinline__ v16bf load_a_frag(const __bf16* base, int ldk, int lane) {
  const int m  = lane & 15;
  const int kh = lane >> 4;
  AFragU f;
  f.v = *(const v16bf*)(base + (size_t)m * ldk + kh * 16);
  return f.v;
}

// B-matrix 32x16 bf16 fragment (K x N): lane n = lane&15 holds one column,
// lanes 0-15 hold K=0..15, lanes 16-31 hold K=16..31 (contiguous per lane).
__device__ __forceinline__ v16bf load_b_frag(const __bf16* colbase, int ldk, int lane) {
  const int n  = lane & 15;
  const int kh = lane >> 4;
  return *(const v16bf*)(colbase + (size_t)n * ldk + kh * 16);
}

__device__ __forceinline__ v8f wmma_bf16(v16bf a, v16bf b, v8f c) {
  return __builtin_amdgcn_wmma_f32_16x16x32_bf16(false, a, false, b, (short)0, c,
                                                 false, false);
}

// ---------------- async global->LDS staging (CDNA5 ISA path, inline asm) ----
// GLOBAL_LOAD_ASYNC_TO_LDS_B128: per-lane 16B memory->LDS, tracked by ASYNCcnt.
// VDST = LDS byte address (low 32 bits of generic shared pointer, per the
// flat-aperture rule); VADDR = 64-bit global addr.
__device__ __forceinline__ void copy16_g2l(const __bf16* g, __bf16* l) {
  const unsigned loff = (unsigned)(size_t)l;
  const unsigned long long ga = (unsigned long long)(size_t)g;
  asm volatile("global_load_async_to_lds_b128 %0, %1, off"
               :
               : "v"(loff), "v"(ga)
               : "memory");
}

__device__ __forceinline__ void wait_stage() {
  asm volatile("s_wait_asynccnt 0" ::: "memory");
  __syncthreads();
}

// ---------------- fp32 -> bf16 convert ----------------
__global__ __launch_bounds__(256) void f32_to_bf16_kernel(const float* __restrict__ src,
                                                          __bf16* __restrict__ dst, int n) {
  int i = blockIdx.x * blockDim.x + threadIdx.x;
  if (i < n) dst[i] = (__bf16)src[i];
}

// ---------------- GEMM: out = A(MxK) * W(NxK)^T ----------------
// mode 1: bf16 out, head-split  [B,H,S,DK]
// mode 2: bf16 out, head-split + transposed [B,H,DK,S]  (for V)
// mode 3: fp32 out, row-major [M,N]  (final projection -> d_out)
__global__ __launch_bounds__(256)
void gemm_bf16_wmma(const __bf16* __restrict__ A, const __bf16* __restrict__ W,
                    void* __restrict__ out, int mode, float alpha) {
  const int lane = threadIdx.x & 31;
  const int wv   = threadIdx.x >> 5;     // 8 waves
  const int wm   = wv >> 2;              // 0..1
  const int wn   = wv & 3;               // 0..3
  const int m0   = blockIdx.y * 128 + wm * 64;   // 64 rows / wave
  const int n0   = blockIdx.x * 128 + wn * 32;   // 32 cols / wave

  v8f c[4][2];
#pragma unroll
  for (int mi = 0; mi < 4; ++mi)
#pragma unroll
    for (int ni = 0; ni < 2; ++ni)
      c[mi][ni] = (v8f){0.f, 0.f, 0.f, 0.f, 0.f, 0.f, 0.f, 0.f};

  for (int k0 = 0; k0 < Dn; k0 += 32) {
    v16bf a[4], b[2];
#pragma unroll
    for (int mi = 0; mi < 4; ++mi)
      a[mi] = load_a_frag(A + (size_t)(m0 + mi * 16) * Dn + k0, Dn, lane);
#pragma unroll
    for (int ni = 0; ni < 2; ++ni)
      b[ni] = load_b_frag(W + (size_t)(n0 + ni * 16) * Dn + k0, Dn, lane);
#pragma unroll
    for (int mi = 0; mi < 4; ++mi)
#pragma unroll
      for (int ni = 0; ni < 2; ++ni)
        c[mi][ni] = wmma_bf16(a[mi], b[ni], c[mi][ni]);
  }

  // C/D layout: VGPR r -> row = r + (lane>>4)*8, col = lane&15
  const int half = lane >> 4;
  const int nl   = lane & 15;
#pragma unroll
  for (int mi = 0; mi < 4; ++mi)
#pragma unroll
    for (int ni = 0; ni < 2; ++ni)
#pragma unroll
      for (int r = 0; r < 8; ++r) {
        const int m = m0 + mi * 16 + r + half * 8;
        const int n = n0 + ni * 16 + nl;
        const float val = c[mi][ni][r] * alpha;
        if (mode == 1) {
          const int b_ = m >> 11, s = m & (Sn - 1), h = n >> 6, dk = n & (DKn - 1);
          ((__bf16*)out)[((size_t)(b_ * Hn + h) * Sn + s) * DKn + dk] = (__bf16)val;
        } else if (mode == 2) {
          const int b_ = m >> 11, s = m & (Sn - 1), h = n >> 6, dk = n & (DKn - 1);
          ((__bf16*)out)[((size_t)(b_ * Hn + h) * DKn + dk) * Sn + s] = (__bf16)val;
        } else {
          ((float*)out)[(size_t)m * Dn + n] = val;
        }
      }
}

// ---------------- Flash attention (causal, online softmax) ----------------
// Q, K: [B,H,S,DK] bf16 (Q pre-scaled by 1/sqrt(DK)); Vt: [B,H,DK,S] bf16.
// O: [B,S,D] bf16.
// Block-synchronous: one block = 128 query rows (8 waves x 16). KV tiles of 32
// are staged into LDS ONCE per block (async-to-LDS, double buffered) and shared
// by all 8 waves; per-wave compute is causally predicated.
__global__ __launch_bounds__(256)
void flash_attn_wmma(const __bf16* __restrict__ Q, const __bf16* __restrict__ Km,
                     const __bf16* __restrict__ Vt, __bf16* __restrict__ O) {
  __shared__ __bf16 ldsK[2][32 * DKn];   // [buf][kv_local][dk]   2 x 4 KB
  __shared__ __bf16 ldsV[2][DKn * 32];   // [buf][dk][s_local]    2 x 4 KB
  __shared__ __bf16 plds[8 * 16 * 32];   // per-wave P tile       8 KB

  const int tid  = threadIdx.x;
  const int lane = tid & 31;
  const int wv   = tid >> 5;
  const int bh   = blockIdx.y;                  // 0..63  (b*H + h)
  const int qbase = blockIdx.x * 128;
  const int q0   = qbase + wv * 16;             // 16 query rows per wave
  const int jmax = qbase + 128;                 // cols needed by last wave

  const __bf16* Qb = Q  + (size_t)bh * Sn * DKn;
  const __bf16* Kb = Km + (size_t)bh * Sn * DKn;
  const __bf16* Vb = Vt + (size_t)bh * DKn * Sn;
  __bf16* myp = plds + wv * (16 * 32);

  // cooperative stage of one 32-wide KV tile into LDS buffer `buf`
  auto stage = [&](int j, int buf) {
    {  // K tile: rows j..j+31 of [S][DK] -> ldsK[buf][32][64]; 256 x 16B
      const int r = tid >> 3;            // 0..31 kv row
      const int c = (tid & 7) * 8;       // dk col, step 8
      copy16_g2l(Kb + (size_t)(j + r) * DKn + c, &ldsK[buf][r * DKn + c]);
    }
    {  // V tile: cols j..j+31 of [DK][S] -> ldsV[buf][64][32]; 256 x 16B
      const int r = tid >> 2;            // 0..63 dk row
      const int c = (tid & 3) * 8;       // s col, step 8
      copy16_g2l(Vb + (size_t)r * Sn + j + c, &ldsV[buf][r * 32 + c]);
    }
  };

  // Resident Q A-fragments (16 x 64 split into two 16x32)
  v16bf qa[2];
#pragma unroll
  for (int kt = 0; kt < 2; ++kt)
    qa[kt] = load_a_frag(Qb + (size_t)q0 * DKn + kt * 32, DKn, lane);

  v8f o[4];
#pragma unroll
  for (int nt = 0; nt < 4; ++nt)
    o[nt] = (v8f){0.f, 0.f, 0.f, 0.f, 0.f, 0.f, 0.f, 0.f};

  float mi[8], li[8];
#pragma unroll
  for (int r = 0; r < 8; ++r) { mi[r] = -__builtin_inff(); li[r] = 0.f; }

  const int half = lane >> 4;
  const int nl   = lane & 15;

  stage(0, 0);
  wait_stage();

  int cur = 0;
  for (int j = 0; j < jmax; j += 32) {
    if (j + 32 < jmax) stage(j + 32, cur ^ 1);   // prefetch next tile (async)

    if (j < q0 + 16) {   // causal: this wave still has work at column tile j
      // ---- scores: S = Q * K^T  (two 16-col tiles) ----
      v8f s[2];
#pragma unroll
      for (int nt = 0; nt < 2; ++nt) {
        s[nt] = (v8f){0.f, 0.f, 0.f, 0.f, 0.f, 0.f, 0.f, 0.f};
#pragma unroll
        for (int kt = 0; kt < 2; ++kt) {
          v16bf kf = load_b_frag(&ldsK[cur][(nt * 16) * DKn + kt * 32], DKn, lane);
          s[nt] = wmma_bf16(qa[kt], kf, s[nt]);
        }
      }

      // ---- causal mask (only near the diagonal) ----
      if (j + 31 > q0) {
#pragma unroll
        for (int nt = 0; nt < 2; ++nt)
#pragma unroll
          for (int r = 0; r < 8; ++r) {
            const int row = q0 + r + half * 8;
            const int col = j + nt * 16 + nl;
            if (col > row) s[nt][r] = -1e9f;
          }
      }

      // ---- online softmax (rows split across 16-lane halves) ----
#pragma unroll
      for (int r = 0; r < 8; ++r) {
        float mx = fmaxf(s[0][r], s[1][r]);
#pragma unroll
        for (int off = 8; off > 0; off >>= 1) mx = fmaxf(mx, __shfl_xor(mx, off, 32));
        const float mnew = fmaxf(mi[r], mx);
        const float corr = __expf(mi[r] - mnew);
        mi[r] = mnew;
        const float p0 = __expf(s[0][r] - mnew);
        const float p1 = __expf(s[1][r] - mnew);
        float rs = p0 + p1;
#pragma unroll
        for (int off = 8; off > 0; off >>= 1) rs += __shfl_xor(rs, off, 32);
        li[r] = li[r] * corr + rs;
#pragma unroll
        for (int nt = 0; nt < 4; ++nt) o[nt][r] *= corr;
        const int m = r + half * 8;
        myp[m * 32 + nl]      = (__bf16)p0;
        myp[m * 32 + 16 + nl] = (__bf16)p1;
      }

      // Per-wave LDS ordering for the P round-trip (no block barrier here)
      asm volatile("s_wait_dscnt 0" ::: "memory");

      // ---- O += P * V ----
      v16bf pa = load_a_frag(myp, 32, lane);
#pragma unroll
      for (int nt = 0; nt < 4; ++nt) {
        v16bf vf = load_b_frag(&ldsV[cur][(nt * 16) * 32], 32, lane);
        o[nt] = wmma_bf16(pa, vf, o[nt]);
      }
      asm volatile("" ::: "memory");
    }

    wait_stage();   // next tile resident; all waves done reading `cur`
    cur ^= 1;
  }

  // ---- epilogue: normalize and store O as bf16 [B,S,D] ----
  float inv[8];
#pragma unroll
  for (int r = 0; r < 8; ++r) inv[r] = 1.0f / li[r];
  const int b_ = bh >> 4, h_ = bh & 15;
#pragma unroll
  for (int nt = 0; nt < 4; ++nt)
#pragma unroll
    for (int r = 0; r < 8; ++r) {
      const int row = q0 + r + half * 8;
      const int d   = h_ * DKn + nt * 16 + nl;
      O[((size_t)b_ * Sn + row) * Dn + d] = (__bf16)(o[nt][r] * inv[r]);
    }
}

// ---------------- host-side launch ----------------
extern "C" void kernel_launch(void* const* d_in, const int* in_sizes, int n_in,
                              void* d_out, int out_size, void* d_ws, size_t ws_size,
                              hipStream_t stream) {
  (void)in_sizes; (void)n_in; (void)out_size; (void)ws_size;

  const float* q_in = (const float*)d_in[0];
  const float* k_in = (const float*)d_in[1];
  const float* v_in = (const float*)d_in[2];
  (void)d_in[3];  // causal mask: implemented analytically
  const float* w_q = (const float*)d_in[4];
  const float* w_k = (const float*)d_in[5];
  const float* w_v = (const float*)d_in[6];
  const float* w_o = (const float*)d_in[7];

  char* ws = (char*)d_ws;
  __bf16* Xbf = (__bf16*)(ws);                         // 16 MB activations (reused)
  __bf16* Wbf = (__bf16*)(ws + ((size_t)16 << 20));    //  2 MB weight (reused)
  __bf16* Qbf = (__bf16*)(ws + ((size_t)18 << 20));    // 16 MB [B,H,S,DK]
  __bf16* Kbf = (__bf16*)(ws + ((size_t)34 << 20));    // 16 MB [B,H,S,DK]
  __bf16* Vtb = (__bf16*)(ws + ((size_t)50 << 20));    // 16 MB [B,H,DK,S]
  __bf16* Obf = Xbf;                                   // reuse after V projection

  const int nAct = Rows * Dn;   // 8388608
  const int nW   = Dn * Dn;     // 1048576
  auto cvt = [&](const float* src, __bf16* dst, int n) {
    f32_to_bf16_kernel<<<(n + 255) / 256, 256, 0, stream>>>(src, dst, n);
  };

  const dim3 ggrid(Dn / 128, Rows / 128);  // (8, 64)
  const float qscale = 0.125f;             // 1/sqrt(DK), folded into Q

  cvt(q_in, Xbf, nAct);
  cvt(w_q, Wbf, nW);
  gemm_bf16_wmma<<<ggrid, 256, 0, stream>>>(Xbf, Wbf, Qbf, 1, qscale);

  cvt(k_in, Xbf, nAct);
  cvt(w_k, Wbf, nW);
  gemm_bf16_wmma<<<ggrid, 256, 0, stream>>>(Xbf, Wbf, Kbf, 1, 1.0f);

  cvt(v_in, Xbf, nAct);
  cvt(w_v, Wbf, nW);
  gemm_bf16_wmma<<<ggrid, 256, 0, stream>>>(Xbf, Wbf, Vtb, 2, 1.0f);

  flash_attn_wmma<<<dim3(Sn / 128, Bn * Hn), 256, 0, stream>>>(Qbf, Kbf, Vtb, Obf);

  cvt(w_o, Wbf, nW);
  gemm_bf16_wmma<<<ggrid, 256, 0, stream>>>(Obf, Wbf, d_out, 3, 1.0f);
}
